// CMXBlock_38895223833288
// MI455X (gfx1250) — matrix operand order repaired and checked by reading, not compiled, not run on hardware
//
#include <hip/hip_runtime.h>
#include <cstdint>

typedef _Float16 f16;
typedef __attribute__((ext_vector_type(16))) _Float16 v16h;
typedef __attribute__((ext_vector_type(8)))  float    v8f;
typedef __attribute__((ext_vector_type(4)))  unsigned int u32x4;
typedef __attribute__((ext_vector_type(8)))  int i32x8;
typedef __attribute__((ext_vector_type(4)))  int i32x4;
typedef uint32_t u32;

static constexpr int Bsz = 8, Cdim = 256, HW = 1024;
static constexpr int NH = 8, HID = 1024;
#define BN_EPS 1e-5f

union U16frag { v16h v; u32 u[8]; f16 h[16]; };

__device__ __forceinline__ v8f wmma32(v16h a, v16h b, v8f c) {
  // D = A(16x32 f16) * B(32x16 f16) + C(16x16 f32)
  return __builtin_amdgcn_wmma_f32_16x16x32_f16(false, a, false, b, (short)0, c,
                                                false, false);
}

// k order this code uses for A (16x32, contiguous, not AMD's table order): element e of v16h, lane-half hi: K = hi*16 + e
__device__ __forceinline__ int a_kmap(int e, int hi) {
  return e + hi * 16;
}
// k order this code uses for B (32x16), matching A: K = hi*16 + e

// ---------------- TDM: 2D tile (f16 elements) global -> LDS, D# per ISA 8.3/8.4
// tile_w contiguous elements per row, tile_h rows, row stride = stride_elems.
// LDS receives the tile packed row-major (tile_h x tile_w halves).
__device__ __forceinline__ void tdm_load_2d_f16(uint32_t lds_off, const void* gptr,
                                                int tensor_w, int tensor_h,
                                                int tile_w, int tile_h,
                                                int stride_elems) {
  uint64_t ga = (uint64_t)(size_t)gptr;
  u32x4 g0;
  g0[0] = 1u;                                   // count=1, user descriptor
  g0[1] = lds_off;                              // lds_addr (bytes)
  g0[2] = (uint32_t)ga;                         // global_addr[31:0]
  g0[3] = (uint32_t)((ga >> 32) & 0x01FFFFFFu)  // global_addr[56:32]
          | (2u << 30);                         // type = 2 ("image")
  i32x8 g1;
  g1[0] = (int)(1u << 16);                      // workgroup_mask=0, data_size=1 (2B)
  g1[1] = (int)(((uint32_t)tensor_w & 0xFFFFu) << 16);            // tensor_dim0[15:0]
  g1[2] = (int)((((uint32_t)tensor_w >> 16) & 0xFFFFu)            // tensor_dim0[31:16]
                | (((uint32_t)tensor_h & 0xFFFFu) << 16));        // tensor_dim1[15:0]
  g1[3] = (int)((((uint32_t)tensor_h >> 16) & 0xFFFFu)            // tensor_dim1[31:16]
                | (((uint32_t)tile_w & 0xFFFFu) << 16));          // tile_dim0
  g1[4] = (int)((uint32_t)tile_h & 0xFFFFu);                      // tile_dim1, tile_dim2=0
  g1[5] = (int)(uint32_t)stride_elems;                            // tensor_dim0_stride[31:0]
  g1[6] = 0;                                                      // stride0[47:32], stride1[15:0]
  g1[7] = 0;                                                      // stride1[47:16]
  i32x4 zz4 = {0, 0, 0, 0};
  i32x8 zz8 = {0, 0, 0, 0, 0, 0, 0, 0};
  // 6-arg variant (clang-23 / therock-10.0 headers)
  __builtin_amdgcn_tensor_load_to_lds(g0, g1, zz4, zz4, zz8, 0);
}

// ---------------------------------------------------------------- f32 -> f16
__global__ void cvt_f16_kernel(const float* __restrict__ src,
                               f16* __restrict__ dst, int n) {
  int i = blockIdx.x * 256 + threadIdx.x;
  if (i < n) dst[i] = (f16)src[i];
}

// ------------------------------------------- fold BN into weights (+ bias)
// W16[o,c] = f16(W[o,c] * inv[c]); bias_out[o] = bias_in[o] + sum_c W[o,c]*sh[c]
__global__ __launch_bounds__(256)
void fold_weights_kernel(const float* __restrict__ W,
                         const float* __restrict__ bias_in,   // may be null
                         const float* __restrict__ g,         // bn params or null
                         const float* __restrict__ bb,
                         const float* __restrict__ mu,
                         const float* __restrict__ var,
                         f16* __restrict__ W16,
                         float* __restrict__ bias_out, int K) {
  int o = blockIdx.x;
  int t = threadIdx.x;
  float part = 0.f;
  for (int c = t; c < K; c += 256) {
    float w = W[(size_t)o * K + c];
    float inv = 1.f, sh = 0.f;
    if (g) {
      inv = g[c] * rsqrtf(var[c] + BN_EPS);
      sh  = bb[c] - mu[c] * inv;
    }
    W16[(size_t)o * K + c] = (f16)(w * inv);
    part += w * sh;
  }
  __shared__ float red[256];
  red[t] = part;
  __syncthreads();
  for (int s = 128; s > 0; s >>= 1) {
    if (t < s) red[t] += red[t + s];
    __syncthreads();
  }
  if (t == 0) bias_out[o] = (bias_in ? bias_in[o] : 0.f) + red[0];
}

// ----------------------------------------------------------- WMMA GEMM
// D[m,n] = sum_k A16[m,k] * B16[b,k,n] + bias[m]  (+ residual, MODE 2)
// MODE 0: plain f16 out [b][M][N]
// MODE 1: head-transposed f16 out [b][m/32][n][m%32]   (Q/V layout, d=32)
// MODE 2: f32 out = acc + bias + res[b][M][N]; optional f16 shadow copy
// A tile is staged into LDS by the Tensor Data Mover (one issue per WG, wave 0);
// B tile is staged transposed by all threads (TDM cannot transpose).
template <int MODE>
__global__ __launch_bounds__(256)
void gemm_wmma_kernel(const f16* __restrict__ A,      // [M,K]
                      const f16* __restrict__ Bm,     // [BATCH,K,N]
                      const float* __restrict__ bias, // [M]
                      const float* __restrict__ res,  // [BATCH,M,N] (MODE 2)
                      f16* __restrict__ out16,
                      float* __restrict__ out32,
                      int M, int K, int N) {
  const int n0 = blockIdx.x * 64;
  const int m0 = blockIdx.y * 64;
  const int b  = blockIdx.z;
  const f16* Bb = Bm + (size_t)b * K * N;

  __shared__ alignas(16) f16 lds_a[64 * 32];   // A tile [row][k] (TDM dest)
  __shared__ alignas(16) f16 lds_bT[64 * 32];  // B tile transposed [n][k]

  const int tid  = threadIdx.x;
  const int lane = tid & 31;
  const int wave = tid >> 5;
  const int wm = (wave >> 1) * 16;   // 4 wave-rows
  const int wn = (wave & 1) * 32;    // 2 wave-cols (each 2x16)
  const int hi = lane >> 4;
  const int ln = lane & 15;
  const uint32_t lds_a_off = (uint32_t)(size_t)(void*)&lds_a[0];

  v8f acc0 = {}, acc1 = {};

  for (int k0 = 0; k0 < K; k0 += 32) {
    // async A tile: 64 rows x 32 halves via Tensor Data Mover (wave 0 issues)
    if (wave == 0)
      tdm_load_2d_f16(lds_a_off, A + (size_t)m0 * K + k0, K, M, 32, 64, K);
    // stage B tile transposed: [n][k]
    for (int p = tid; p < 32 * 64; p += 256) {
      int k = p >> 6, n = p & 63;
      lds_bT[n * 32 + k] = Bb[(size_t)(k0 + k) * N + n0 + n];
    }
    // prefetch next B tile (global_prefetch_b8)
    if (k0 + 32 < K)
      __builtin_prefetch(&Bb[(size_t)(k0 + 32 + (tid & 31)) * N + n0 + (tid >> 5) * 8],
                         0, 1);
    if (wave == 0) __builtin_amdgcn_s_wait_tensorcnt(0);
    __syncthreads();

    U16frag af, b0, b1;
#pragma unroll
    for (int p = 0; p < 8; ++p) {
      int kb = a_kmap(2 * p, hi);
      af.u[p] = *(const u32*)(&lds_a[(wm + ln) * 32 + kb]);
    }
#pragma unroll
    for (int p = 0; p < 8; ++p) {
      int kb = hi * 16 + 2 * p;
      b0.u[p] = *(const u32*)(&lds_bT[(wn + ln) * 32 + kb]);
      b1.u[p] = *(const u32*)(&lds_bT[(wn + 16 + ln) * 32 + kb]);
    }
    acc0 = wmma32(af.v, b0.v, acc0);
    acc1 = wmma32(af.v, b1.v, acc1);
    __syncthreads();
  }

  // epilogue: lane holds col=ln (per 16-block), rows r + 8*hi
#pragma unroll
  for (int r = 0; r < 8; ++r) {
    int mrow = m0 + wm + r + 8 * hi;
    float bs = bias[mrow];
    float v0 = acc0[r] + bs;
    float v1 = acc1[r] + bs;
    int nc0 = n0 + wn + ln;
    int nc1 = nc0 + 16;
    if (MODE == 0) {
      size_t base = ((size_t)b * M + mrow) * N;
      out16[base + nc0] = (f16)v0;
      out16[base + nc1] = (f16)v1;
    } else if (MODE == 1) {
      int h = mrow >> 5, dc = mrow & 31;
      size_t base = (size_t)(b * NH + h) * N;
      out16[(base + nc0) * 32 + dc] = (f16)v0;
      out16[(base + nc1) * 32 + dc] = (f16)v1;
    } else {
      size_t base = ((size_t)b * M + mrow) * N;
      float y0 = v0 + res[base + nc0];
      float y1 = v1 + res[base + nc1];
      out32[base + nc0] = y0;
      out32[base + nc1] = y1;
      if (out16) {
        out16[base + nc0] = (f16)y0;
        out16[base + nc1] = (f16)y1;
      }
    }
  }
}

// ------------------------------------------- flash attention, 1 wave per WG
// Q,V: [B,NH,HW,32] f16; K: [B,NH,32,HW] f16; out: [B,C,HW] f16
__global__ __launch_bounds__(32)
void attn_wmma_kernel(const f16* __restrict__ Q, const f16* __restrict__ Km,
                      const f16* __restrict__ V, const float* __restrict__ temp,
                      f16* __restrict__ out) {
  const int r0 = blockIdx.x * 16;
  const int h  = blockIdx.y;
  const int b  = blockIdx.z;
  const int lane = threadIdx.x & 31;
  const int hi = lane >> 4, ln = lane & 15;
  const float ts = temp[h];

  __shared__ alignas(16) float ldsP[16 * 32];

  const f16* Qb = Q  + ((size_t)(b * NH + h) * HW + r0) * 32;
  const f16* Kb = Km + (size_t)(b * NH + h) * 32 * HW;
  const f16* Vb = V  + (size_t)(b * NH + h) * HW * 32;

  U16frag qa;
#pragma unroll
  for (int e = 0; e < 16; ++e) qa.h[e] = Qb[ln * 32 + a_kmap(e, hi)];

  float mi[8], li[8];
  v8f o0 = {}, o1 = {};
#pragma unroll
  for (int r = 0; r < 8; ++r) { mi[r] = -1e30f; li[r] = 0.f; }

  for (int j0 = 0; j0 < HW; j0 += 32) {
    U16frag kb0, kb1;
#pragma unroll
    for (int e = 0; e < 16; ++e) {
      int kk = hi * 16 + e;
      kb0.h[e] = Kb[(size_t)kk * HW + j0 + ln];
      kb1.h[e] = Kb[(size_t)kk * HW + j0 + 16 + ln];
    }
    v8f z = {};
    v8f s0 = wmma32(qa.v, kb0.v, z);
    v8f s1 = wmma32(qa.v, kb1.v, z);

    float p0[8], p1[8];
#pragma unroll
    for (int r = 0; r < 8; ++r) {
      float a0 = s0[r] * ts, a1 = s1[r] * ts;
      float mx = fmaxf(a0, a1);
      for (int m = 1; m < 16; m <<= 1) mx = fmaxf(mx, __shfl_xor(mx, m, 32));
      float mn = fmaxf(mi[r], mx);
      float alpha = expf(mi[r] - mn);
      float e0 = expf(a0 - mn), e1 = expf(a1 - mn);
      float rs = e0 + e1;
      for (int m = 1; m < 16; m <<= 1) rs += __shfl_xor(rs, m, 32);
      li[r] = li[r] * alpha + rs;
      mi[r] = mn;
      o0[r] *= alpha;
      o1[r] *= alpha;
      p0[r] = e0;
      p1[r] = e1;
    }
    // C-layout -> A-layout transpose of P through LDS
#pragma unroll
    for (int r = 0; r < 8; ++r) {
      ldsP[(r + 8 * hi) * 32 + ln]      = p0[r];
      ldsP[(r + 8 * hi) * 32 + 16 + ln] = p1[r];
    }
    __syncthreads();
    U16frag pa;
#pragma unroll
    for (int e = 0; e < 16; ++e) pa.h[e] = (f16)ldsP[ln * 32 + a_kmap(e, hi)];
    __syncthreads();

    U16frag vb0, vb1;
#pragma unroll
    for (int e = 0; e < 16; ++e) {
      int kk = hi * 16 + e;
      vb0.h[e] = Vb[(size_t)(j0 + kk) * 32 + ln];
      vb1.h[e] = Vb[(size_t)(j0 + kk) * 32 + 16 + ln];
    }
    o0 = wmma32(pa.v, vb0.v, o0);
    o1 = wmma32(pa.v, vb1.v, o1);
  }

#pragma unroll
  for (int r = 0; r < 8; ++r) {
    int row = r + 8 * hi;
    float inv = 1.f / li[r];
    int c0 = h * 32 + ln, c1 = c0 + 16;
    out[((size_t)b * Cdim + c0) * HW + r0 + row] = (f16)(o0[r] * inv);
    out[((size_t)b * Cdim + c1) * HW + r0 + row] = (f16)(o1[r] * inv);
  }
}

// ------------------------------ depthwise 3x3 (SAME) + exact GELU, f16->f16
__global__ void dwconv_gelu_kernel(const f16* __restrict__ src,
                                   const float* __restrict__ w, // [HID,1,3,3]
                                   f16* __restrict__ dst) {
  int idx = blockIdx.x * 256 + threadIdx.x;  // B*HID*HW = 8M
  int x = idx & 31, y = (idx >> 5) & 31;
  int c = (idx >> 10) & (HID - 1);
  int b = idx >> 20;
  const f16* s = src + (((size_t)b * HID + c) << 10);
  const float* wc = w + c * 9;
  float acc = 0.f;
#pragma unroll
  for (int dy = -1; dy <= 1; ++dy)
#pragma unroll
    for (int dx = -1; dx <= 1; ++dx) {
      int yy = y + dy, xx = x + dx;
      if (yy >= 0 && yy < 32 && xx >= 0 && xx < 32)
        acc += wc[(dy + 1) * 3 + (dx + 1)] * (float)s[yy * 32 + xx];
    }
  float g = 0.5f * acc * (1.f + erff(acc * 0.70710678118654752f));
  dst[idx] = (f16)g;
}

// =======================================================================
extern "C" void kernel_launch(void* const* d_in, const int* in_sizes, int n_in,
                              void* d_out, int out_size, void* d_ws,
                              size_t ws_size, hipStream_t stream) {
  const float* x1     = (const float*)d_in[0];
  const float* x2     = (const float*)d_in[1];
  const float* bn1_g  = (const float*)d_in[2];
  const float* bn1_b  = (const float*)d_in[3];
  const float* bn1_m  = (const float*)d_in[4];
  const float* bn1_v  = (const float*)d_in[5];
  const float* q_w    = (const float*)d_in[6];
  const float* k_w    = (const float*)d_in[7];
  const float* v_w    = (const float*)d_in[8];
  const float* temp   = (const float*)d_in[9];
  const float* proj_w = (const float*)d_in[10];
  const float* proj_b = (const float*)d_in[11];
  const float* bn2_g  = (const float*)d_in[12];
  const float* bn2_b  = (const float*)d_in[13];
  const float* bn2_m  = (const float*)d_in[14];
  const float* bn2_v  = (const float*)d_in[15];
  const float* fc1_w  = (const float*)d_in[16];
  const float* fc1_b  = (const float*)d_in[17];
  const float* dw_w   = (const float*)d_in[18];
  const float* fc2_w  = (const float*)d_in[19];
  const float* fc2_b  = (const float*)d_in[20];
  (void)in_sizes; (void)n_in; (void)out_size; (void)ws_size;

  char* ws = (char*)d_ws;
  size_t off = 0;
  auto take = [&](size_t bytes) -> char* {
    char* p = ws + off;
    off += (bytes + 255) & ~(size_t)255;
    return p;
  };

  const size_t ACT = (size_t)Bsz * Cdim * HW;       // 2M elems
  const size_t ACTH = (size_t)Bsz * HID * HW;       // 8M elems

  f16* x1h    = (f16*)take(ACT * 2);
  f16* x2h    = (f16*)take(ACT * 2);
  f16* qbuf   = (f16*)take(ACT * 2);   // [B,NH,HW,32]
  f16* kbuf   = (f16*)take(ACT * 2);   // [B,NH,32,HW] (== plain [B,C,HW])
  f16* vbuf   = (f16*)take(ACT * 2);   // [B,NH,HW,32]
  f16* attn16 = (f16*)take(ACT * 2);   // [B,C,HW]
  float* y1f  = (float*)take(ACT * 4); // x1 + attn residual, f32
  f16* y1h    = (f16*)take(ACT * 2);
  f16* h16    = (f16*)take(ACTH * 2);  // fc1 out
  f16* g16    = (f16*)take(ACTH * 2);  // dw+gelu out
  f16* qw16   = (f16*)take((size_t)Cdim * Cdim * 2);
  f16* kw16   = (f16*)take((size_t)Cdim * Cdim * 2);
  f16* vw16   = (f16*)take((size_t)Cdim * Cdim * 2);
  f16* pw16   = (f16*)take((size_t)Cdim * Cdim * 2);
  f16* f1w16  = (f16*)take((size_t)HID * Cdim * 2);
  f16* f2w16  = (f16*)take((size_t)Cdim * HID * 2);
  float* qb   = (float*)take(Cdim * 4);
  float* kb   = (float*)take(Cdim * 4);
  float* vb   = (float*)take(Cdim * 4);
  float* pb   = (float*)take(Cdim * 4);
  float* f1b  = (float*)take(HID * 4);
  float* f2b  = (float*)take(Cdim * 4);

  // 1) convert raw activations to f16 (BN is folded into weights)
  cvt_f16_kernel<<<(int)(ACT / 256), 256, 0, stream>>>(x1, x1h, (int)ACT);
  cvt_f16_kernel<<<(int)(ACT / 256), 256, 0, stream>>>(x2, x2h, (int)ACT);

  // 2) fold BN1 into q/k/v, BN2 into fc1; convert proj/fc2
  fold_weights_kernel<<<Cdim, 256, 0, stream>>>(q_w, nullptr, bn1_g, bn1_b,
                                                bn1_m, bn1_v, qw16, qb, Cdim);
  fold_weights_kernel<<<Cdim, 256, 0, stream>>>(k_w, nullptr, bn1_g, bn1_b,
                                                bn1_m, bn1_v, kw16, kb, Cdim);
  fold_weights_kernel<<<Cdim, 256, 0, stream>>>(v_w, nullptr, bn1_g, bn1_b,
                                                bn1_m, bn1_v, vw16, vb, Cdim);
  fold_weights_kernel<<<Cdim, 256, 0, stream>>>(proj_w, proj_b, nullptr, nullptr,
                                                nullptr, nullptr, pw16, pb, Cdim);
  fold_weights_kernel<<<HID, 256, 0, stream>>>(fc1_w, fc1_b, bn2_g, bn2_b,
                                               bn2_m, bn2_v, f1w16, f1b, Cdim);
  fold_weights_kernel<<<Cdim, 256, 0, stream>>>(fc2_w, fc2_b, nullptr, nullptr,
                                                nullptr, nullptr, f2w16, f2b, HID);

  // 3) Q/K/V projections (WMMA GEMMs, TDM-staged A tiles)
  dim3 gQKV(HW / 64, Cdim / 64, Bsz);
  gemm_wmma_kernel<1><<<gQKV, 256, 0, stream>>>(qw16, x1h, qb, nullptr, qbuf,
                                                nullptr, Cdim, Cdim, HW);
  gemm_wmma_kernel<0><<<gQKV, 256, 0, stream>>>(kw16, x2h, kb, nullptr, kbuf,
                                                nullptr, Cdim, Cdim, HW);
  gemm_wmma_kernel<1><<<gQKV, 256, 0, stream>>>(vw16, x2h, vb, nullptr, vbuf,
                                                nullptr, Cdim, Cdim, HW);

  // 4) flash attention, one wave per (b, head, 16-query tile)
  attn_wmma_kernel<<<dim3(HW / 16, NH, Bsz), 32, 0, stream>>>(qbuf, kbuf, vbuf,
                                                              temp, attn16);

  // 5) proj + residual: y1 = x1 + proj(attn) (f32 + f16 shadow)
  gemm_wmma_kernel<2><<<gQKV, 256, 0, stream>>>(pw16, attn16, pb, x1, y1h, y1f,
                                                Cdim, Cdim, HW);

  // 6) fc1 (BN2 folded) -> h16 [B,HID,HW]
  dim3 gFC1(HW / 64, HID / 64, Bsz);
  gemm_wmma_kernel<0><<<gFC1, 256, 0, stream>>>(f1w16, y1h, f1b, nullptr, h16,
                                                nullptr, HID, Cdim, HW);

  // 7) depthwise 3x3 + exact GELU
  dwconv_gelu_kernel<<<(int)(ACTH / 256), 256, 0, stream>>>(h16, dw_w, g16);

  // 8) fc2 + residual -> d_out (f32)
  gemm_wmma_kernel<2><<<gQKV, 256, 0, stream>>>(f2w16, g16, f2b, y1f, nullptr,
                                                (float*)d_out, Cdim, HID, HW);
}
